// QuantizedLinear_78323023610033
// MI455X (gfx1250) — matrix operand order repaired and checked by reading, not compiled, not run on hardware
//
#include <hip/hip_runtime.h>

typedef __attribute__((ext_vector_type(16))) __bf16 v16bf;
typedef __attribute__((ext_vector_type(8)))  __bf16 v8bf;
typedef __attribute__((ext_vector_type(4)))  __bf16 v4bf;
typedef __attribute__((ext_vector_type(2)))  __bf16 v2bf;
typedef __attribute__((ext_vector_type(8)))  float  v8f;
typedef __attribute__((ext_vector_type(4)))  float  v4f;
typedef __attribute__((ext_vector_type(2)))  float  v2f;

__constant__ float NF4_LUT[16] = {
    -1.0f, -0.6962f, -0.5251f, -0.3949f, -0.2844f, -0.1848f, -0.0911f, 0.0f,
     0.0796f, 0.1609f, 0.2461f, 0.3379f, 0.4407f, 0.5626f, 0.723f, 1.0f};

constexpr int OUT_F  = 11008;
constexpr int IN_F   = 4096;
constexpr int MROWS  = 64;
constexpr int NT     = 32;      // N-tile per workgroup
constexpr int KT     = 32;      // K-step (matches 16x16x32 WMMA)
constexpr int KP     = KT + 8;  // padded LDS row stride (80 B: kills 4-way conflicts)
constexpr int KSPLIT = 4;       // K split across blockIdx.y for wave concurrency
constexpr int KCHUNK = IN_F / KSPLIT;   // 1024
constexpr int NSTEP  = KCHUNK / KT;     // 32

// ---- Kernel 1: x fp32 -> bf16 (512 KB, stays L2-resident) ----
__global__ void xcvt_kernel(const float* __restrict__ x,
                            unsigned short* __restrict__ xb) {
  int t = blockIdx.x * blockDim.x + threadIdx.x;     // one float4 per thread
  v4f  v = ((const v4f*)x)[t];
  v4bf b = __builtin_convertvector(v, v4bf);         // native pk cvt if available
  ((uint2*)xb)[t] = __builtin_bit_cast(uint2, b);
}

// ---- Kernel 2: out[m][n] = bias[n] (atomic accumulation target) ----
__global__ void bias_init_kernel(const float* __restrict__ bias,
                                 float* __restrict__ out) {
  int t = blockIdx.x * blockDim.x + threadIdx.x;     // 64*11008 threads
  out[t] = bias[t % OUT_F];
}

// ---- Kernel 3: fused NF4 dequant + bf16 WMMA GEMM (K-split, atomic add) ----
__global__ __launch_bounds__(256) void nf4_gemm_kernel(
    const unsigned short* __restrict__ xb,     // [64, 4096] bf16
    const int*            __restrict__ idx,    // [11008*4096] codes 0..15
    const float*          __restrict__ scales, // [11008*64]
    float*                __restrict__ out)    // [64, 11008] (pre-filled w/ bias)
{
  __shared__ unsigned short wlds[2][NT * KP]; // ping-pong bf16 W tile, padded rows
  __shared__ float lut[16];

  const int t = threadIdx.x;
  if (t < 16) lut[t] = NF4_LUT[t];

  const int nbase = blockIdx.x * NT;
  const int kofs  = blockIdx.y * KCHUNK;
  const int w     = t >> 5;        // wave id 0..7 (wave32)
  const int lane  = t & 31;
  const int l16   = lane & 15;
  const int khalf = lane >> 4;

  const int mrow  = (w & 3) * 16 + l16;            // A row for this lane
  const int nsub  = w >> 2;                        // which 16-wide N subtile
  const int nglob = nbase + nsub * 16 + l16;       // B/C column for this lane

  // dequant assignment: each thread handles 4 consecutive k of one tile row
  const int drow = t >> 3;                         // 0..31 (n offset in tile)
  const int dk   = (t & 7) * 4;                    // 0,4,...,28
  const long long idx_row  = (long long)(nbase + drow) * IN_F;
  const long long scal_row = (long long)(nbase + drow) * (IN_F / 64);

  v8f acc = {};
  int p = 0;

  // scale register: a 64-wide quant block covers two KT=32 steps
  float sc = scales[scal_row + (kofs >> 6)];

  // prologue: dequant step 0 into buffer 0
  {
    int4 q = *(const int4*)(idx + idx_row + kofs + dk);
    __syncthreads();                               // lut ready
    v2f f01 = {lut[q.x & 15] * sc, lut[q.y & 15] * sc};
    v2f f23 = {lut[q.z & 15] * sc, lut[q.w & 15] * sc};
    uint2 o;
    o.x = __builtin_bit_cast(unsigned, __builtin_convertvector(f01, v2bf));
    o.y = __builtin_bit_cast(unsigned, __builtin_convertvector(f23, v2bf));
    *(uint2*)&wlds[0][drow * KP + dk] = o;
  }
  __syncthreads();

  for (int s = 0; s < NSTEP; ++s) {
    const int kbase = kofs + s * KT;

    // A fragment from global bf16 x (L2 hit). ISA 16-bit A 16x32 layout:
    // lane elems 0..7 = K khalf*8.., elems 8..15 = K 16+khalf*8..
    const unsigned short* ap = xb + (long long)mrow * IN_F + kbase + khalf * 8;
    v8bf alo = *(const v8bf*)ap;
    v8bf ahi = *(const v8bf*)(ap + 16);
    v16bf a = __builtin_shufflevector(alo, ahi, 0,1,2,3,4,5,6,7,8,9,10,11,12,13,14,15);

    // B fragment from LDS: lane = col n, elems 0..7 = K khalf*8.., elems 8..15 = K 16+khalf*8..
    const unsigned short* bp = &wlds[p][(nsub * 16 + l16) * KP + khalf * 8];
    v8bf blo = *(const v8bf*)bp;
    v8bf bhi = *(const v8bf*)(bp + 16);
    v16bf b = __builtin_shufflevector(blo, bhi, 0,1,2,3,4,5,6,7,8,9,10,11,12,13,14,15);

    // overlap: dequant next K-step into the other buffer
    if (s + 1 < NSTEP) {
      const int kb2 = kbase + KT;
      int4 q = *(const int4*)(idx + idx_row + kb2 + dk);
      if ((kb2 & 63) == 0)                      // entered a new quant block
        sc = scales[scal_row + (kb2 >> 6)];
      v2f f01 = {lut[q.x & 15] * sc, lut[q.y & 15] * sc};
      v2f f23 = {lut[q.z & 15] * sc, lut[q.w & 15] * sc};
      uint2 o;
      o.x = __builtin_bit_cast(unsigned, __builtin_convertvector(f01, v2bf));
      o.y = __builtin_bit_cast(unsigned, __builtin_convertvector(f23, v2bf));
      *(uint2*)&wlds[p ^ 1][drow * KP + dk] = o;
    }

    acc = __builtin_amdgcn_wmma_f32_16x16x32_bf16(
        /*neg_a=*/false, a, /*neg_b=*/false, b,
        /*c_mod=*/(short)0, acc, /*reuse_a=*/false, /*reuse_b=*/false);

    __syncthreads();
    p ^= 1;
  }

  // epilogue: C VGPR r -> rows (r | r+8) per lane half; accumulate partials
  const int mtop = (w & 3) * 16 + khalf * 8;
#pragma unroll
  for (int r = 0; r < 8; ++r)
    unsafeAtomicAdd(&out[(long long)(mtop + r) * OUT_F + nglob], acc[r]);
}

// ---- Kernel 4: sparse outlier correction: C[:,n] += x[:,k]*(val - dequant) ----
__global__ void outlier_fix_kernel(const float* __restrict__ x,
                                   const int*   __restrict__ idx,
                                   const float* __restrict__ scales,
                                   const int*   __restrict__ oidx,
                                   const float* __restrict__ oval,
                                   float*       __restrict__ out,
                                   int n_out)
{
  int t = blockIdx.x * blockDim.x + threadIdx.x;
  int o = t >> 6;                 // outlier id
  int m = t & 63;                 // batch row
  if (o >= n_out) return;
  long long p = (long long)(unsigned)oidx[o];
  int n = (int)(p / IN_F);
  int k = (int)(p % IN_F);
  float dq    = NF4_LUT[idx[p] & 15] * scales[p >> 6];
  float delta = oval[o] - dq;
  unsafeAtomicAdd(&out[(long long)m * OUT_F + n],
                  x[(long long)m * IN_F + k] * delta);
}

extern "C" void kernel_launch(void* const* d_in, const int* in_sizes, int n_in,
                              void* d_out, int out_size, void* d_ws, size_t ws_size,
                              hipStream_t stream) {
  const float* x       = (const float*)d_in[0];   // [64,4096]
  const int*   indices = (const int*)d_in[1];     // [704512,64] == flat W codes
  const float* scales  = (const float*)d_in[2];   // [704512]
  const int*   oidx    = (const int*)d_in[3];     // [45088]
  const float* oval    = (const float*)d_in[4];   // [45088]
  const float* bias    = (const float*)d_in[5];   // [11008]
  float*       out     = (float*)d_out;           // [64,11008]
  unsigned short* xb   = (unsigned short*)d_ws;   // bf16 x, 512 KB

  // 1) x -> bf16 (L2-resident operand)
  {
    int nvec = (MROWS * IN_F) / 4;
    xcvt_kernel<<<nvec / 256, 256, 0, stream>>>(x, xb);
  }
  // 2) out = broadcast(bias)
  bias_init_kernel<<<(MROWS * OUT_F) / 256, 256, 0, stream>>>(bias, out);
  // 3) fused dequant + WMMA GEMM, K split 4-ways for wave concurrency
  {
    dim3 grid(OUT_F / NT, KSPLIT);
    nf4_gemm_kernel<<<grid, 256, 0, stream>>>(xb, indices, scales, out);
  }
  // 4) sparse outlier correction
  {
    int n_out = in_sizes[3];
    int tot   = n_out * 64;
    outlier_fix_kernel<<<(tot + 255) / 256, 256, 0, stream>>>(
        x, indices, scales, oidx, oval, out, n_out);
  }
}
